// TemporalDecayAttention_81226421502324
// MI455X (gfx1250) — matrix operand order repaired and checked by reading, not compiled, not run on hardware
//
#include <hip/hip_runtime.h>
#include <hip/hip_bf16.h>

typedef __bf16 bfloat;
typedef __attribute__((ext_vector_type(16))) __bf16 v16bf;
typedef __attribute__((ext_vector_type(8)))  __bf16 v8bf;
typedef __attribute__((ext_vector_type(4)))  __bf16 v4bf;
typedef __attribute__((ext_vector_type(8)))  float  v8f;

#define WMMA_BF16(a, b, c) \
  __builtin_amdgcn_wmma_f32_16x16x32_bf16(false, (a), false, (b), (short)0, (c), false, false)

// Problem constants (from reference: B=2, L=4096, D=512, H=8, Hd=64)
constexpr int   LSEQ   = 4096;
constexpr int   DMODEL = 512;
constexpr int   MROWS  = 2 * LSEQ;          // 8192
constexpr float QSCALE = 0.125f;            // 1/sqrt(64)
constexpr float LOG2E  = 1.4426950408889634f;
constexpr float LOG2DECAY = -0.07400058144377693f; // log2(0.95)

// ---------------- fragment loaders ----------------
// A-matrix 16x32 bf16 layout: lanes 0-15 = rows, lane half selects
// K chunks {0..15} vs {16..31} (contiguous, matching fragB).
__device__ __forceinline__ v16bf fragA(const bfloat* row_k0, int hi) {
  union { v16bf f; v8bf h[2]; } u;
  u.h[0] = *(const v8bf*)(row_k0 + hi * 16);
  u.h[1] = *(const v8bf*)(row_k0 + hi * 16 + 8);
  return u.f;
}
// B-matrix 32x16 bf16 layout: lane = column; lanes 0-15 hold K=0..15 (vgpr v ->
// K=2v,2v+1), lanes 16-31 hold K=16..31. One 32-byte contiguous run per lane.
__device__ __forceinline__ v16bf fragB(const bfloat* row_k0, int hi) {
  union { v16bf f; v8bf h[2]; } u;
  u.h[0] = *(const v8bf*)(row_k0 + hi * 16);
  u.h[1] = *(const v8bf*)(row_k0 + hi * 16 + 8);
  return u.f;
}
// pack two f32 -> bf16x2 dword
__device__ __forceinline__ unsigned int pkbf(float a, float b) {
  union { __bf16 h[2]; unsigned int u; } t;
  t.h[0] = (__bf16)a;
  t.h[1] = (__bf16)b;
  return t.u;
}

// ---------------- fp32 -> bf16 convert ----------------
__global__ void cvt_f32_bf16(const float* __restrict__ in, bfloat* __restrict__ out, int n4) {
  int i = blockIdx.x * blockDim.x + threadIdx.x;
  if (i >= n4) return;
  float4 v = ((const float4*)in)[i];
  v4bf o;
  o[0] = (bfloat)v.x; o[1] = (bfloat)v.y; o[2] = (bfloat)v.z; o[3] = (bfloat)v.w;
  *(v4bf*)(out + i * 4) = o;
}

// ---------------- GEMM: out = A(bf16)[8192x512] @ W(bf16)[512x512]^T + bias ----------------
// One wave computes a 64x64 tile: 4x4 WMMA accumulators, K loop of 16 steps x 16 WMMAs.
// mode 0: bf16 store to [B,H,L,64], value = (acc+bias)*scale      (Q with scale, K with 1)
// mode 1: bf16 store transposed to [B,H,64,L] (V^T), packed b128 stores
// mode 2: fp32 store row-major [8192,512], value = acc+bias       (final projection)
__global__ void __launch_bounds__(128)
gemm512_wmma(const bfloat* __restrict__ A, const bfloat* __restrict__ W,
             const float* __restrict__ bias, bfloat* __restrict__ outb,
             float* __restrict__ outf, int mode, float scale) {
  const int lane = threadIdx.x & 31;
  const int lr = lane & 15, hi = lane >> 4;
  const int wid = (blockIdx.x * blockDim.x + threadIdx.x) >> 5; // 0..1023
  const int m0 = (wid >> 3) * 64;
  const int n0 = (wid & 7) * 64;

  v8f acc[4][4] = {};

  for (int k0 = 0; k0 < DMODEL; k0 += 32) {
    v16bf af[4], bf[4];
#pragma unroll
    for (int mt = 0; mt < 4; mt++)
      af[mt] = fragA(A + (size_t)(m0 + mt * 16 + lr) * DMODEL + k0, hi);
#pragma unroll
    for (int nt = 0; nt < 4; nt++)
      bf[nt] = fragB(W + (size_t)(n0 + nt * 16 + lr) * DMODEL + k0, hi);
#pragma unroll
    for (int mt = 0; mt < 4; mt++)
#pragma unroll
      for (int nt = 0; nt < 4; nt++)
        acc[mt][nt] = WMMA_BF16(af[mt], bf[nt], acc[mt][nt]);
  }

  float bv[4];
#pragma unroll
  for (int nt = 0; nt < 4; nt++) bv[nt] = bias[n0 + nt * 16 + lr];

  if (mode == 0) {
#pragma unroll
    for (int mt = 0; mt < 4; mt++) {
      int mbase = m0 + mt * 16 + hi * 8;       // C layout: vgpr r -> row mbase+r
      int b = mbase >> 12, l0 = mbase & 4095;
#pragma unroll
      for (int nt = 0; nt < 4; nt++) {
        int col = n0 + nt * 16 + lr;
        int h = col >> 6, hd = col & 63;
        size_t base = ((size_t)(b * 8 + h) * LSEQ) * 64 + (size_t)hd;
#pragma unroll
        for (int r = 0; r < 8; r++)
          outb[base + (size_t)(l0 + r) * 64] = (bfloat)((acc[mt][nt][r] + bv[nt]) * scale);
      }
    }
  } else if (mode == 1) {
#pragma unroll
    for (int mt = 0; mt < 4; mt++) {
      int mbase = m0 + mt * 16 + hi * 8;
      int b = mbase >> 12, l0 = mbase & 4095;
#pragma unroll
      for (int nt = 0; nt < 4; nt++) {
        int col = n0 + nt * 16 + lr;
        int h = col >> 6, hd = col & 63;
        size_t addr = ((size_t)(b * 8 + h) * 64 + hd) * LSEQ + l0;
        v8bf pk;
#pragma unroll
        for (int r = 0; r < 8; r++) pk[r] = (bfloat)(acc[mt][nt][r] + bv[nt]);
        *(v8bf*)(outb + addr) = pk;             // 8 consecutive l values -> b128 store
      }
    }
  } else {
#pragma unroll
    for (int mt = 0; mt < 4; mt++) {
      int mbase = m0 + mt * 16 + hi * 8;
#pragma unroll
      for (int nt = 0; nt < 4; nt++) {
        int col = n0 + nt * 16 + lr;
#pragma unroll
        for (int r = 0; r < 8; r++)
          outf[(size_t)(mbase + r) * DMODEL + col] = acc[mt][nt][r] + bv[nt];
      }
    }
  }
}

// ---------------- Flash attention with temporal-decay mask (transposed scores) ----------------
// Q,K: [16][4096][64] bf16 (Q pre-scaled by 1/8). Vt: [16][64][4096] bf16.
// Oa: [8192][512] bf16 (head-merged, feeds final GEMM).
// One wave owns 32 query columns; streams 128 key-blocks of 32 keys.
// S^T = K.Q^T so the softmax (key) axis is vertical in-lane: per-lane m/l scalars,
// single cross-half shuffle per reduction, no LDS, no barriers.
__global__ void __launch_bounds__(128)
attn_decay_wmma(const bfloat* __restrict__ Q, const bfloat* __restrict__ K,
                const bfloat* __restrict__ Vt, bfloat* __restrict__ Oa) {
  const int lane = threadIdx.x & 31;
  const int lr = lane & 15, hi = lane >> 4;
  const int wid = blockIdx.x * 4 + (threadIdx.x >> 5); // 0..2047
  const int bh = wid >> 7;                             // b*8+h
  const int q0 = (wid & 127) * 32;

  const bfloat* Qh = Q + (size_t)bh * LSEQ * 64;
  const bfloat* Kh = K + (size_t)bh * LSEQ * 64;
  const bfloat* Vh = Vt + (size_t)bh * 64 * LSEQ;

  // resident Q fragments in B layout (lane = query column i, contiguous d)
  v16bf qb[2][2];
#pragma unroll
  for (int it = 0; it < 2; it++)
#pragma unroll
    for (int kt = 0; kt < 2; kt++)
      qb[it][kt] = fragB(Qh + (size_t)(q0 + it * 16 + lr) * 64 + kt * 32, hi);

  v8f acc[4][2] = {};                 // acc[dt][it] = out^T tile (rows d, cols i)
  float mcur[2] = {-3.0e38f, -3.0e38f};
  float lcur[2] = {0.0f, 0.0f};
  const float fi0 = (float)(q0 + lr);
  const float fi1 = (float)(q0 + 16 + lr);

  for (int kb = 0; kb < LSEQ / 32; kb++) {
    const int kbase = kb * 32;

    // K in A layout (lane = key row j, contiguous d)
    v16bf ka[2][2];
#pragma unroll
    for (int jt = 0; jt < 2; jt++)
#pragma unroll
      for (int kt = 0; kt < 2; kt++)
        ka[jt][kt] = fragA(Kh + (size_t)(kbase + jt * 16 + lr) * 64 + kt * 32, hi);

    // V^T rows as A-frags (lane = d row, contiguous j) — issue loads early
    v16bf va[4];
#pragma unroll
    for (int dt = 0; dt < 4; dt++)
      va[dt] = fragA(Vh + (size_t)(dt * 16 + lr) * LSEQ + kbase, hi);

    // S^T = K @ Q^T : St[jt][it], C layout rows j = kbase+jt*16+hi*8+r, col i = lane
    v8f St[2][2] = {};
#pragma unroll
    for (int jt = 0; jt < 2; jt++)
#pragma unroll
      for (int it = 0; it < 2; it++)
#pragma unroll
        for (int kt = 0; kt < 2; kt++)
          St[jt][it] = WMMA_BF16(ka[jt][kt], qb[it][kt], St[jt][it]);

    float alpha[2];
#pragma unroll
    for (int it = 0; it < 2; it++) {
      const float fi = it ? fi1 : fi0;
      // temporal decay mask (multiplies scores pre-softmax)
#pragma unroll
      for (int jt = 0; jt < 2; jt++)
#pragma unroll
        for (int r = 0; r < 8; r++) {
          float fj = (float)(kbase + jt * 16 + hi * 8 + r);
          St[jt][it][r] *= exp2f(LOG2DECAY * fabsf(fi - fj));
        }
      // column max over j: vertical in-lane + one cross-half shuffle
      float t = St[0][it][0];
#pragma unroll
      for (int r = 1; r < 8; r++) t = fmaxf(t, St[0][it][r]);
#pragma unroll
      for (int r = 0; r < 8; r++) t = fmaxf(t, St[1][it][r]);
      t = fmaxf(t, __shfl_xor(t, 16));
      float newm = fmaxf(mcur[it], t);
      alpha[it] = exp2f((mcur[it] - newm) * LOG2E);
      mcur[it] = newm;
      // p = exp(s - m); column sum the same way
      float rs = 0.0f;
#pragma unroll
      for (int jt = 0; jt < 2; jt++)
#pragma unroll
        for (int r = 0; r < 8; r++) {
          float p = exp2f((St[jt][it][r] - newm) * LOG2E);
          St[jt][it][r] = p;
          rs += p;
        }
      rs += __shfl_xor(rs, 16);
      lcur[it] = lcur[it] * alpha[it] + rs;
      // rescale accumulator (alpha is a single per-lane scalar)
#pragma unroll
      for (int dt = 0; dt < 4; dt++)
#pragma unroll
        for (int r = 0; r < 8; r++)
          acc[dt][it][r] *= alpha[it];
    }

    // P^T B-fragments straight from registers: pack bf16 pairs, swap cross-half
    // k-chunks with one shfl_xor(16) per dword, select own/partner per half.
#pragma unroll
    for (int it = 0; it < 2; it++) {
      union { v16bf f; unsigned int u[8]; } uf;
#pragma unroll
      for (int k2 = 0; k2 < 4; k2++) {
        unsigned int own0 = pkbf(St[0][it][2 * k2], St[0][it][2 * k2 + 1]); // j: jt0 pairs
        unsigned int own1 = pkbf(St[1][it][2 * k2], St[1][it][2 * k2 + 1]); // j: jt1 pairs
        unsigned int oth0 = (unsigned int)__shfl_xor((int)own0, 16);
        unsigned int oth1 = (unsigned int)__shfl_xor((int)own1, 16);
        uf.u[k2]     = hi ? oth1 : own0; // k = hi*16 + {0..7}
        uf.u[k2 + 4] = hi ? own1 : oth0; // k = hi*16 + {8..15}
      }
      // acc^T += V^T @ P^T
#pragma unroll
      for (int dt = 0; dt < 4; dt++)
        acc[dt][it] = WMMA_BF16(va[dt], uf.f, acc[dt][it]);
    }
  }

  // normalize; out^T layout gives 8 consecutive d per (dt,it) -> packed b128 stores
  const int b = bh >> 3, h = bh & 7;
#pragma unroll
  for (int it = 0; it < 2; it++) {
    float inv = 1.0f / lcur[it];
    int row = b * LSEQ + q0 + it * 16 + lr;
#pragma unroll
    for (int dt = 0; dt < 4; dt++) {
      v8bf pk;
#pragma unroll
      for (int r = 0; r < 8; r++) pk[r] = (bfloat)(acc[dt][it][r] * inv);
      *(v8bf*)(Oa + (size_t)row * DMODEL + h * 64 + dt * 16 + hi * 8) = pk;
    }
  }
}

extern "C" void kernel_launch(void* const* d_in, const int* in_sizes, int n_in,
                              void* d_out, int out_size, void* d_ws, size_t ws_size,
                              hipStream_t stream) {
  const float* x  = (const float*)d_in[0];
  const float* Wq = (const float*)d_in[1];
  const float* bq = (const float*)d_in[2];
  const float* Wk = (const float*)d_in[3];
  const float* bk = (const float*)d_in[4];
  const float* Wv = (const float*)d_in[5];
  const float* bv = (const float*)d_in[6];
  const float* Wo = (const float*)d_in[7];
  const float* bo = (const float*)d_in[8];
  float* out = (float*)d_out;

  const size_t NX = (size_t)MROWS * DMODEL;  // 4194304
  const size_t NW = (size_t)DMODEL * DMODEL; // 262144

  bfloat* ws  = (bfloat*)d_ws;
  bfloat* xb  = ws;
  bfloat* wqb = xb + NX;
  bfloat* wkb = wqb + NW;
  bfloat* wvb = wkb + NW;
  bfloat* wob = wvb + NW;
  bfloat* Qb  = wob + NW;   // [16][4096][64]
  bfloat* Kb  = Qb + NX;
  bfloat* Vtb = Kb + NX;    // [16][64][4096]
  bfloat* Ob  = Vtb + NX;   // [8192][512]

  // fp32 -> bf16
  cvt_f32_bf16<<<(int)(NX / 4 / 256), 256, 0, stream>>>(x, xb, (int)(NX / 4));
  cvt_f32_bf16<<<(int)(NW / 4 / 256), 256, 0, stream>>>(Wq, wqb, (int)(NW / 4));
  cvt_f32_bf16<<<(int)(NW / 4 / 256), 256, 0, stream>>>(Wk, wkb, (int)(NW / 4));
  cvt_f32_bf16<<<(int)(NW / 4 / 256), 256, 0, stream>>>(Wv, wvb, (int)(NW / 4));
  cvt_f32_bf16<<<(int)(NW / 4 / 256), 256, 0, stream>>>(Wo, wob, (int)(NW / 4));

  // projections: 1024 wave-tiles -> 256 blocks of 4 waves
  gemm512_wmma<<<256, 128, 0, stream>>>(xb, wqb, bq, Qb, nullptr, 0, QSCALE);
  gemm512_wmma<<<256, 128, 0, stream>>>(xb, wkb, bk, Kb, nullptr, 0, 1.0f);
  gemm512_wmma<<<256, 128, 0, stream>>>(xb, wvb, bv, Vtb, nullptr, 1, 1.0f);

  // flash attention: 2048 waves -> 512 blocks of 4 waves
  attn_decay_wmma<<<512, 128, 0, stream>>>(Qb, Kb, Vtb, Ob);

  // output projection (fp32 result)
  gemm512_wmma<<<256, 128, 0, stream>>>(Ob, wob, bo, nullptr, out, 2, 1.0f);
}
